// CrossAttention_ResTransformer_55078660604025
// MI455X (gfx1250) — matrix operand order repaired and checked by reading, not compiled, not run on hardware
//
#include <hip/hip_runtime.h>

typedef __attribute__((ext_vector_type(16))) __bf16 v16bf;
typedef __attribute__((ext_vector_type(8)))  float  v8f;

#define N_TOK   4096
#define DIM_IN  514
#define OUT_D   256
#define QKV_LD  768
#define FEAT_D  1542   // 3*DIM_IN

__device__ __forceinline__ int a_kloc(int e, int hi) {
    int v = e >> 1, half = e & 1;
    return ((v & 4) ? 16 : 0) + hi * 8 + ((v & 3) << 1) + half;
}

union BfPack2 { __bf16 h[2]; unsigned int  u; };
union BfPack4 { __bf16 h[4]; unsigned long long u; };

// ---------------------------------------------------------------- convert ----
// n must be a multiple of 4 (true for all our tensors)
__global__ __launch_bounds__(256)
void f32_to_bf16x4_kernel(const float* __restrict__ src, __bf16* __restrict__ dst, int n4) {
    int i = blockIdx.x * 256 + threadIdx.x;
    if (i >= n4) return;
    float4 f = ((const float4*)src)[i];
    BfPack4 p;
    p.h[0] = (__bf16)f.x; p.h[1] = (__bf16)f.y; p.h[2] = (__bf16)f.z; p.h[3] = (__bf16)f.w;
    ((unsigned long long*)dst)[i] = p.u;
}

// ---------------------------------------------------------------- GEMM -------
// C[M,N] = A[M,K](bf16) * B[K,N](bf16) + bias + resid   (f32 out, col-offset via C ptr/ldc)
// Requires: M % 128 == 0, lda/ldb even (true for 514/512/768).
__global__ __launch_bounds__(256)
void gemm_bf16_kernel(const __bf16* __restrict__ A, int lda,
                      const __bf16* __restrict__ B, int ldb,
                      float* __restrict__ C, int ldc,
                      const float* __restrict__ bias,
                      const float* __restrict__ resid, int ldr,
                      int M, int N, int K) {
    constexpr int BM = 128, BN = 128, BK = 32;
    __shared__ __bf16 As[BM][BK + 2];   // row-major A tile
    __shared__ __bf16 Bs[BN][BK + 8];   // TRANSPOSED B tile: [n][k], k contiguous

    const int tid  = threadIdx.x;
    const int lane = tid & 31;
    const int wv   = tid >> 5;
    const int ln   = lane & 15;
    const int hi   = lane >> 4;
    const int bm   = blockIdx.x * BM;
    const int bn   = blockIdx.y * BN;
    const int wm   = (wv >> 1) * 32;   // 4 waves down M
    const int wn   = (wv & 1) * 64;    // 2 waves across N

    const bool colFull = (bn + BN <= N);   // block-uniform

    const v8f zero8 = {0.f,0.f,0.f,0.f,0.f,0.f,0.f,0.f};
    v8f acc[2][4];
    #pragma unroll
    for (int i = 0; i < 2; ++i)
        #pragma unroll
        for (int j = 0; j < 4; ++j) acc[i][j] = zero8;

    const int ar = tid >> 1, akh = (tid & 1) * 16;       // A staging role
    const int bkr = tid >> 3, bcb = (tid & 7) * 16;      // B staging role

    for (int k0 = 0; k0 < K; k0 += BK) {
        const bool fullK = (k0 + BK <= K);               // block-uniform
        // ---- stage A: one 16-elem half-row per thread
        {
            const __bf16* Arow = A + (size_t)(bm + ar) * lda + k0 + akh;
            if (fullK) {
                const unsigned int* src = (const unsigned int*)Arow;     // 4B aligned
                unsigned int* dst = (unsigned int*)&As[ar][akh];         // 4B aligned
                #pragma unroll
                for (int i = 0; i < 8; ++i) dst[i] = src[i];
            } else {
                #pragma unroll
                for (int i = 0; i < 16; ++i) {
                    int k = k0 + akh + i;
                    __bf16 vv = A[(size_t)(bm + ar) * lda + (k < K ? k : 0)]; // clamped addr
                    As[ar][akh + i] = (k < K) ? vv : (__bf16)0.0f;            // cndmask
                }
            }
        }
        // ---- stage B transposed: one k-row x 16 cols per thread
        {
            const int gk = k0 + bkr;
            if (fullK && colFull) {
                const unsigned int* src = (const unsigned int*)(B + (size_t)gk * ldb + bn + bcb);
                #pragma unroll
                for (int i = 0; i < 8; ++i) {
                    BfPack2 p; p.u = src[i];
                    Bs[bcb + 2 * i][bkr]     = p.h[0];
                    Bs[bcb + 2 * i + 1][bkr] = p.h[1];
                }
            } else {
                #pragma unroll
                for (int i = 0; i < 16; ++i) {
                    int c = bn + bcb + i;
                    bool ok = (gk < K) && (c < N);
                    __bf16 vv = B[(size_t)(gk < K ? gk : 0) * ldb + (c < N ? c : 0)];
                    Bs[bcb + i][bkr] = ok ? vv : (__bf16)0.0f;
                }
            }
        }
        __syncthreads();

        v16bf af[2], bf[4];
        #pragma unroll
        for (int sm = 0; sm < 2; ++sm) {
            int rm = wm + sm * 16 + ln;
            #pragma unroll
            for (int e = 0; e < 16; ++e) af[sm][e] = As[rm][hi * 16 + e];
        }
        #pragma unroll
        for (int sn = 0; sn < 4; ++sn) {
            int cn = wn + sn * 16 + ln;
            #pragma unroll
            for (int e = 0; e < 16; ++e) bf[sn][e] = Bs[cn][hi * 16 + e];
        }
        #pragma unroll
        for (int sm = 0; sm < 2; ++sm)
            #pragma unroll
            for (int sn = 0; sn < 4; ++sn)
                acc[sm][sn] = __builtin_amdgcn_wmma_f32_16x16x32_bf16(
                    false, af[sm], false, bf[sn], (short)0, acc[sm][sn], false, false);
        __syncthreads();
    }

    // epilogue: lane holds col n=ln, rows hi*8+v
    #pragma unroll
    for (int sn = 0; sn < 4; ++sn) {
        int gcol = bn + wn + sn * 16 + ln;
        if (gcol >= N) continue;
        float bval = bias ? bias[gcol] : 0.f;
        #pragma unroll
        for (int sm = 0; sm < 2; ++sm) {
            #pragma unroll
            for (int v = 0; v < 8; ++v) {
                int grow = bm + wm + sm * 16 + hi * 8 + v;
                float val = acc[sm][sn][v] + bval;
                if (resid) val += resid[(size_t)grow * ldr + gcol];
                C[(size_t)grow * ldc + gcol] = val;
            }
        }
    }
}

// ------------------------------------------------------------ attention -----
// flash attention, unscaled softmax(Q K^T) V ; qkv f32 [3][4096][768], out f32 [6][4096][256]
__global__ __launch_bounds__(256)
void attn_kernel(const float* __restrict__ qkv, float* __restrict__ oatt) {
    const int g     = blockIdx.y;
    const int qmod  = ((1 | (2 << 3) | (0 << 6) | (2 << 9) | (0 << 12) | (1 << 15)) >> (3 * g)) & 7;
    const int kvmod = g >> 1;

    const float* Qg = qkv + (size_t)qmod  * N_TOK * QKV_LD;          // cols [0,256)
    const float* Kg = qkv + (size_t)kvmod * N_TOK * QKV_LD + 256;    // cols [256,512)
    const float* Vg = qkv + (size_t)kvmod * N_TOK * QKV_LD + 512;    // cols [512,768)

    __shared__ __bf16 Klds[32][264];     // keys x d      (d contiguous)
    __shared__ __bf16 VldsT[256][40];    // d x keys      (keys contiguous)
    __shared__ __bf16 Plds[8][16][40];   // per-wave P    (keys contiguous)

    const int tid  = threadIdx.x;
    const int lane = tid & 31;
    const int wv   = tid >> 5;
    const int ln   = lane & 15;
    const int hi   = lane >> 4;
    const int q0   = blockIdx.x * 128 + wv * 16;

    // Q fragments held in registers (A layout); pair elements are k-contiguous -> float2 loads
    v16bf qf[8];
    {
        const float* qrow = Qg + (size_t)(q0 + ln) * QKV_LD;
        #pragma unroll
        for (int kk = 0; kk < 8; ++kk)
            #pragma unroll
            for (int v = 0; v < 8; ++v) {
                int kl = hi * 16 + (v << 1);
                float2 q2 = *(const float2*)&qrow[kk * 32 + kl];
                qf[kk][2 * v]     = (__bf16)q2.x;
                qf[kk][2 * v + 1] = (__bf16)q2.y;
            }
    }

    const v8f zero8 = {0.f,0.f,0.f,0.f,0.f,0.f,0.f,0.f};
    v8f O[16];
    #pragma unroll
    for (int t = 0; t < 16; ++t) O[t] = zero8;
    float mrow[8], lrow[8];
    #pragma unroll
    for (int v = 0; v < 8; ++v) { mrow[v] = -__builtin_inff(); lrow[v] = 0.f; }

    for (int key0 = 0; key0 < N_TOK; key0 += 32) {
        // stage K (row-major, packed b64 stores) and V (transposed), float4 global loads
        #pragma unroll
        for (int rep = 0; rep < 8; ++rep) {
            int i  = tid + rep * 256;          // 0 .. 2047
            int kr = i >> 6, d4 = (i & 63) << 2;
            const float4 kf = *(const float4*)&Kg[(size_t)(key0 + kr) * QKV_LD + d4];
            const float4 vf = *(const float4*)&Vg[(size_t)(key0 + kr) * QKV_LD + d4];
            BfPack4 p;
            p.h[0] = (__bf16)kf.x; p.h[1] = (__bf16)kf.y;
            p.h[2] = (__bf16)kf.z; p.h[3] = (__bf16)kf.w;
            *(unsigned long long*)&Klds[kr][d4] = p.u;          // 8B-aligned ds_store_b64
            VldsT[d4 + 0][kr] = (__bf16)vf.x;
            VldsT[d4 + 1][kr] = (__bf16)vf.y;
            VldsT[d4 + 2][kr] = (__bf16)vf.z;
            VldsT[d4 + 3][kr] = (__bf16)vf.w;
        }
        if (key0 + 32 < N_TOK)
            __builtin_prefetch(&Kg[(size_t)(key0 + 32 + (tid >> 3)) * QKV_LD], 0, 0);
        __syncthreads();

        // S = Q K^T  (16 queries x 32 keys, f32)
        v8f S0 = zero8, S1 = zero8;
        #pragma unroll
        for (int kk = 0; kk < 8; ++kk) {
            v16bf b0, b1;
            #pragma unroll
            for (int e = 0; e < 16; ++e) {
                b0[e] = Klds[ln][kk * 32 + hi * 16 + e];
                b1[e] = Klds[16 + ln][kk * 32 + hi * 16 + e];
            }
            S0 = __builtin_amdgcn_wmma_f32_16x16x32_bf16(false, qf[kk], false, b0, (short)0, S0, false, false);
            S1 = __builtin_amdgcn_wmma_f32_16x16x32_bf16(false, qf[kk], false, b1, (short)0, S1, false, false);
        }

        // online softmax (rows live in 16-lane halves; xor 1,2,4,8 reduces a row)
        #pragma unroll
        for (int v = 0; v < 8; ++v) {
            float rm = fmaxf(S0[v], S1[v]);
            #pragma unroll
            for (int o = 1; o < 16; o <<= 1) rm = fmaxf(rm, __shfl_xor(rm, o, 32));
            float nm = fmaxf(mrow[v], rm);
            float sc = __expf(mrow[v] - nm);
            mrow[v] = nm;
            #pragma unroll
            for (int t = 0; t < 16; ++t) O[t][v] *= sc;
            float p0 = __expf(S0[v] - nm);
            float p1 = __expf(S1[v] - nm);
            float ps = p0 + p1;
            #pragma unroll
            for (int o = 1; o < 16; o <<= 1) ps += __shfl_xor(ps, o, 32);
            lrow[v] = lrow[v] * sc + ps;
            Plds[wv][hi * 8 + v][ln]      = (__bf16)p0;
            Plds[wv][hi * 8 + v][16 + ln] = (__bf16)p1;
        }
        __syncthreads();

        // O += P V   (A = P 16x32, B = V 32xd)
        v16bf pf;
        #pragma unroll
        for (int e = 0; e < 16; ++e) pf[e] = Plds[wv][ln][hi * 16 + e];
        #pragma unroll
        for (int t = 0; t < 16; ++t) {
            v16bf bv;
            #pragma unroll
            for (int e = 0; e < 16; ++e) bv[e] = VldsT[t * 16 + ln][hi * 16 + e];
            O[t] = __builtin_amdgcn_wmma_f32_16x16x32_bf16(false, pf, false, bv, (short)0, O[t], false, false);
        }
        __syncthreads();
    }

    float* Og = oatt + (size_t)g * N_TOK * OUT_D;
    #pragma unroll
    for (int v = 0; v < 8; ++v) {
        float inv = 1.0f / lrow[v];
        int row = q0 + hi * 8 + v;
        #pragma unroll
        for (int t = 0; t < 16; ++t)
            Og[(size_t)row * OUT_D + t * 16 + ln] = O[t][v] * inv;
    }
}

// --------------------------------------------------------------- pack t_r ---
// tr[3][4096][512] = concat(out_a, out_b) + concat(k, k)   (bf16), 4 elems/thread
__global__ __launch_bounds__(256)
void pack_tr_kernel(const float* __restrict__ oatt, const float* __restrict__ qkv,
                    __bf16* __restrict__ tr) {
    int i4 = blockIdx.x * 256 + threadIdx.x;           // index into groups of 4
    if (i4 >= 3 * N_TOK * 512 / 4) return;
    int idx = i4 << 2;
    int mod = idx / (N_TOK * 512);
    int rem = idx - mod * (N_TOK * 512);
    int row = rem >> 9;
    int c   = rem & 511;                               // multiple of 4
    int cc  = c & 255;
    int ga  = ((2 | (0 << 2) | (1 << 4)) >> (2 * mod)) & 3;   // {out21,out12,out13}
    int gb  = ((4 | (5 << 3) | (3 << 6)) >> (3 * mod)) & 7;   // {out31,out32,out23}
    int gsel = (c < 256) ? ga : gb;
    const float4 kv = *(const float4*)&qkv[(size_t)mod * N_TOK * QKV_LD + (size_t)row * QKV_LD + 256 + cc];
    const float4 ov = *(const float4*)&oatt[(size_t)gsel * N_TOK * OUT_D + (size_t)row * OUT_D + cc];
    BfPack4 p;
    p.h[0] = (__bf16)(ov.x + kv.x);
    p.h[1] = (__bf16)(ov.y + kv.y);
    p.h[2] = (__bf16)(ov.z + kv.z);
    p.h[3] = (__bf16)(ov.w + kv.w);
    *(unsigned long long*)&tr[idx] = p.u;
}

// ------------------------------------------------------------- LN + head ----
__global__ __launch_bounds__(256)
void ln_head_kernel(const float* __restrict__ atten,
                    const float* __restrict__ lg, const float* __restrict__ lb,
                    const float* __restrict__ Wh, const float* __restrict__ bh,
                    float* __restrict__ out) {
    int row  = blockIdx.x;
    const float* x = atten + (size_t)row * FEAT_D;
    int tid = threadIdx.x, lane = tid & 31, wv = tid >> 5;
    __shared__ float red0[8], red1[8];

    float s = 0.f, ss = 0.f;
    for (int c = tid; c < FEAT_D; c += 256) { float v = x[c]; s += v; ss += v * v; }
    #pragma unroll
    for (int o = 1; o < 32; o <<= 1) { s += __shfl_xor(s, o, 32); ss += __shfl_xor(ss, o, 32); }
    if (lane == 0) { red0[wv] = s; red1[wv] = ss; }
    __syncthreads();
    float S = 0.f, SS = 0.f;
    #pragma unroll
    for (int w = 0; w < 8; ++w) { S += red0[w]; SS += red1[w]; }
    const float invn = 1.0f / (float)FEAT_D;
    float mu  = S * invn;
    float var = SS * invn - mu * mu;
    float rstd = rsqrtf(var + 1e-5f);

    float a0 = 0.f, a1 = 0.f;
    for (int c = tid; c < FEAT_D; c += 256) {
        float nv = (x[c] - mu) * rstd * lg[c] + lb[c];
        a0 += nv * Wh[2 * c];
        a1 += nv * Wh[2 * c + 1];
    }
    #pragma unroll
    for (int o = 1; o < 32; o <<= 1) { a0 += __shfl_xor(a0, o, 32); a1 += __shfl_xor(a1, o, 32); }
    __syncthreads();
    if (lane == 0) { red0[wv] = a0; red1[wv] = a1; }
    __syncthreads();
    if (tid == 0) {
        float A0 = 0.f, A1 = 0.f;
        #pragma unroll
        for (int w = 0; w < 8; ++w) { A0 += red0[w]; A1 += red1[w]; }
        out[(size_t)row * 2 + 0] = A0 + bh[0];
        out[(size_t)row * 2 + 1] = A1 + bh[1];
    }
}

// --------------------------------------------------------------- launch -----
extern "C" void kernel_launch(void* const* d_in, const int* in_sizes, int n_in,
                              void* d_out, int out_size, void* d_ws, size_t ws_size,
                              hipStream_t stream) {
    (void)in_sizes; (void)n_in; (void)out_size; (void)ws_size;

    const float* t_in[3] = { (const float*)d_in[0], (const float*)d_in[1], (const float*)d_in[2] };
    const float* Wqkv[3] = { (const float*)d_in[3], (const float*)d_in[5], (const float*)d_in[7] };
    const float* bqkv[3] = { (const float*)d_in[4], (const float*)d_in[6], (const float*)d_in[8] };
    const float* Wo[3]   = { (const float*)d_in[9], (const float*)d_in[11], (const float*)d_in[13] };
    const float* bo[3]   = { (const float*)d_in[10], (const float*)d_in[12], (const float*)d_in[14] };
    const float* ln_g = (const float*)d_in[15];
    const float* ln_b = (const float*)d_in[16];
    const float* Wh   = (const float*)d_in[17];
    const float* bh   = (const float*)d_in[18];

    char* ws = (char*)d_ws;
    size_t off = 0;
    auto take = [&](size_t bytes) { char* p = ws + off; off = (off + bytes + 255) & ~(size_t)255; return p; };

    __bf16* tb[3];  for (int i = 0; i < 3; ++i) tb[i]  = (__bf16*)take((size_t)N_TOK * DIM_IN * 2);
    __bf16* wqb[3]; for (int i = 0; i < 3; ++i) wqb[i] = (__bf16*)take((size_t)DIM_IN * QKV_LD * 2);
    __bf16* wob[3]; for (int i = 0; i < 3; ++i) wob[i] = (__bf16*)take((size_t)512 * DIM_IN * 2);
    float*  qkvf = (float*)take((size_t)3 * N_TOK * QKV_LD * 4);
    float*  oatt = (float*)take((size_t)6 * N_TOK * OUT_D * 4);
    __bf16* trb  = (__bf16*)take((size_t)3 * N_TOK * 512 * 2);

    // 0) f32 -> bf16 conversions (all sizes divisible by 4)
    for (int i = 0; i < 3; ++i) {
        int n4 = (N_TOK * DIM_IN) / 4;
        f32_to_bf16x4_kernel<<<(n4 + 255) / 256, 256, 0, stream>>>(t_in[i], tb[i], n4);
    }
    for (int i = 0; i < 3; ++i) {
        int n4 = (DIM_IN * QKV_LD) / 4;
        f32_to_bf16x4_kernel<<<(n4 + 255) / 256, 256, 0, stream>>>(Wqkv[i], wqb[i], n4);
    }
    for (int i = 0; i < 3; ++i) {
        int n4 = (512 * DIM_IN) / 4;
        f32_to_bf16x4_kernel<<<(n4 + 255) / 256, 256, 0, stream>>>(Wo[i], wob[i], n4);
    }

    // 1) QKV projections: [4096,514] @ [514,768] + bias
    for (int i = 0; i < 3; ++i)
        gemm_bf16_kernel<<<dim3(N_TOK / 128, QKV_LD / 128), 256, 0, stream>>>(
            tb[i], DIM_IN, wqb[i], QKV_LD,
            qkvf + (size_t)i * N_TOK * QKV_LD, QKV_LD,
            bqkv[i], nullptr, 0, N_TOK, QKV_LD, DIM_IN);

    // 2) six flash cross-attentions
    attn_kernel<<<dim3(N_TOK / 128, 6), 256, 0, stream>>>(qkvf, oatt);

    // 3) residual pack -> bf16 t_r
    pack_tr_kernel<<<(3 * N_TOK * 512 / 4) / 256, 256, 0, stream>>>(oatt, qkvf, trb);

    // 4) output projections + bias + residual, written straight into feat region of d_out
    float* feat = (float*)d_out + (size_t)N_TOK * 2;
    for (int i = 0; i < 3; ++i)
        gemm_bf16_kernel<<<dim3(N_TOK / 128, (DIM_IN + 127) / 128), 256, 0, stream>>>(
            trb + (size_t)i * N_TOK * 512, 512, wob[i], DIM_IN,
            feat + (size_t)i * DIM_IN, FEAT_D,
            bo[i], t_in[i], DIM_IN, N_TOK, DIM_IN, 512);

    // 5) layernorm + classifier head
    ln_head_kernel<<<N_TOK, 256, 0, stream>>>(feat, ln_g, ln_b, Wh, bh, (float*)d_out);
}
